// App_Classifier_22471268893055
// MI455X (gfx1250) — matrix operand order repaired and checked by reading, not compiled, not run on hardware
//
#include <hip/hip_runtime.h>
#include <hip/hip_bf16.h>

// ---------------------------------------------------------------------------
// CDNA5 (gfx1250, wave32) GCN pipeline.
// GEMMs via v_wmma_f32_16x16x32_bf16 (fp32 accumulate) with a branch-free
// inner loop: A rows are zero-padded to a multiple of 32 in the workspace
// (so no K guards -> global_load_b128), and B is pre-packed into the wave32
// WMMA B-fragment layout in bf16 (one 32B load per K-chunk).
// Edge message passing via global_atomic_add_f32 (L2-resident: all node
// buffers fit in the 192MB L2).
// ---------------------------------------------------------------------------

typedef __attribute__((ext_vector_type(16))) __bf16 v16bf;
typedef __attribute__((ext_vector_type(8)))  float  v8f;

// -------------------------- utility kernels --------------------------------

__global__ void zero_f32(float* __restrict__ p, long n) {
  long i = (long)blockIdx.x * blockDim.x + threadIdx.x;
  long stride = (long)gridDim.x * blockDim.x;
  for (; i < n; i += stride) p[i] = 0.0f;
}

__global__ void degree_kernel(const int* __restrict__ src, const int* __restrict__ dst,
                              float* __restrict__ deg_out, float* __restrict__ deg_in, int E) {
  int i = blockIdx.x * blockDim.x + threadIdx.x;
  if (i < E) {
    atomicAdd(&deg_out[src[i]], 1.0f);
    atomicAdd(&deg_in[dst[i]], 1.0f);
  }
}

// in-place: d[i] = rsqrt(max(d[i], 1))
__global__ void rsqrt_clip_kernel(float* __restrict__ d, int n) {
  int i = blockIdx.x * blockDim.x + threadIdx.x;
  if (i < n) d[i] = rsqrtf(fmaxf(d[i], 1.0f));
}

// agg[dst[e]] += x[src[e]] * inv_sqrt_out[src[e]]
// x rows have stride 100 (D=100); agg rows have padded stride LDA.
__global__ void edge_scatter_kernel(const float* __restrict__ x,
                                    const float* __restrict__ inv_sqrt_out,
                                    const int* __restrict__ src, const int* __restrict__ dst,
                                    float* __restrict__ agg, int E, int lda) {
  int gid = blockIdx.x * blockDim.x + threadIdx.x;
  int total = E * 25;                       // 100 / 4 = 25 float4 per edge
  if (gid >= total) return;
  int e = gid / 25;
  int c = (gid - e * 25) * 4;
  int s = src[e];
  int d = dst[e];
  float sc = inv_sqrt_out[s];
  const float4 v = *reinterpret_cast<const float4*>(x + (size_t)s * 100 + c);
  float* o = agg + (size_t)d * lda + c;
  atomicAdd(o + 0, v.x * sc);
  atomicAdd(o + 1, v.y * sc);
  atomicAdd(o + 2, v.z * sc);
  atomicAdd(o + 3, v.w * sc);
}

__global__ void count_kernel(const int* __restrict__ gidm, float* __restrict__ cnt, int N) {
  int i = blockIdx.x * blockDim.x + threadIdx.x;
  if (i < N) atomicAdd(&cnt[gidm[i]], 1.0f);
}

// pooled[graph_id[node]] += h2[node]; h2 stride 200, pooled padded stride LDA
__global__ void pool_scatter_kernel(const float* __restrict__ h,
                                    const int* __restrict__ gidm,
                                    float* __restrict__ pooled, int N, int lda) {
  int gid = blockIdx.x * blockDim.x + threadIdx.x;
  int total = N * 50;                       // 200 / 4 = 50 float4 per node
  if (gid >= total) return;
  int node = gid / 50;
  int c = (gid - node * 50) * 4;
  int g = gidm[node];
  const float4 v = *reinterpret_cast<const float4*>(h + (size_t)node * 200 + c);
  float* o = pooled + (size_t)g * lda + c;
  atomicAdd(o + 0, v.x);
  atomicAdd(o + 1, v.y);
  atomicAdd(o + 2, v.z);
  atomicAdd(o + 3, v.w);
}

// in-place: cnt[i] = 1 / max(cnt[i], 1)  -> per-row scale for final GEMM
__global__ void inv_count_kernel(float* __restrict__ cnt, int G) {
  int i = blockIdx.x * blockDim.x + threadIdx.x;
  if (i < G) cnt[i] = 1.0f / fmaxf(cnt[i], 1.0f);
}

// --------------------- B-fragment pre-pack (bf16) ---------------------------
// Packs W[K,N] (row-major f32) into wave32 WMMA B-fragment layout, zero-padding
// K up to Kchunks*32 and N up to Ntiles*16.  Per ISA 7.12.2: lane holds column
// n0+(lane&15); element e -> K = kc*32 + (lane>>4)*8 + (e<8 ? e : e+8).
// out layout: [Ntiles][Kchunks][32 lanes][16 bf16]  (32 bytes per lane-chunk)
__global__ void pack_B_kernel(const float* __restrict__ W, __bf16* __restrict__ out,
                              int K, int N, int Kchunks, int total) {
  int tid = blockIdx.x * blockDim.x + threadIdx.x;
  if (tid >= total) return;                 // total = Ntiles*Kchunks*32
  int lane = tid & 31;
  int t = tid >> 5;
  int kc = t % Kchunks;
  int nt = t / Kchunks;
  int l15 = lane & 15, hi = lane >> 4;
  int bcol = nt * 16 + l15;
  int kbase = kc * 32 + hi * 8;
  v16bf v;
#pragma unroll
  for (int e = 0; e < 16; ++e) {
    int kb = kbase + (e < 8 ? e : e + 8);
    float val = (bcol < N && kb < K) ? W[(size_t)kb * N + bcol] : 0.0f;
    v[e] = (__bf16)val;
  }
  *reinterpret_cast<v16bf*>(out + (size_t)tid * 16) = v;
}

// ---------------------------- WMMA GEMM ------------------------------------
// Y[M,N] = (rowscale ? diag(rowscale) : I) * A[M, Kchunks*32] @ B + bias
// A has row stride lda with rows valid/zero out to Kchunks*32 (caller pads).
// B pre-packed by pack_B_kernel.  One wave per 16x16 output tile.
// A fragment (ISA 7.12.2, 16-bit A 16x32): lane row = m0+(lane&15);
//   elems 0-7 -> K k0+(lane>>4)*8+0..7, elems 8-15 -> same +16.
// D (f32 16x16): elem r -> row m0 + r + (lane>>4)*8, col n0 + (lane&15).
__global__ __launch_bounds__(128)
void gemm_bias_wmma_kernel(const float* __restrict__ A,
                           const float* __restrict__ rowscale,
                           const __bf16* __restrict__ Bpack,
                           const float* __restrict__ bias,
                           float* __restrict__ Y,
                           int lda, int Kchunks, int N, int Ntiles, int totalTiles) {
  int lane = threadIdx.x & 31;
  int wave = (blockIdx.x * blockDim.x + threadIdx.x) >> 5;
  if (wave >= totalTiles) return;               // uniform per wave
  int mt = wave / Ntiles;
  int nt = wave - mt * Ntiles;
  int m0 = mt << 4, n0 = nt << 4;

  int l15 = lane & 15;
  int hi  = lane >> 4;
  int arow = m0 + l15;                          // always < M (M % 16 == 0)

  float rs = rowscale ? rowscale[arow] : 1.0f;
  const float* Arow = A + (size_t)arow * lda + hi * 8;
  const v16bf* Bp = reinterpret_cast<const v16bf*>(Bpack)
                    + (size_t)nt * Kchunks * 32 + lane;

  v8f acc = {};
  for (int kc = 0; kc < Kchunks; ++kc) {
    const float* ap = Arow + kc * 32;
    float4 a0 = *reinterpret_cast<const float4*>(ap);       // K +0..3
    float4 a1 = *reinterpret_cast<const float4*>(ap + 4);   // K +4..7
    float4 a2 = *reinterpret_cast<const float4*>(ap + 16);  // K +16..19
    float4 a3 = *reinterpret_cast<const float4*>(ap + 20);  // K +20..23
    v16bf b = Bp[(size_t)kc * 32];
    v16bf a;
    a[0]  = (__bf16)(a0.x * rs); a[1]  = (__bf16)(a0.y * rs);
    a[2]  = (__bf16)(a0.z * rs); a[3]  = (__bf16)(a0.w * rs);
    a[4]  = (__bf16)(a1.x * rs); a[5]  = (__bf16)(a1.y * rs);
    a[6]  = (__bf16)(a1.z * rs); a[7]  = (__bf16)(a1.w * rs);
    a[8]  = (__bf16)(a2.x * rs); a[9]  = (__bf16)(a2.y * rs);
    a[10] = (__bf16)(a2.z * rs); a[11] = (__bf16)(a2.w * rs);
    a[12] = (__bf16)(a3.x * rs); a[13] = (__bf16)(a3.y * rs);
    a[14] = (__bf16)(a3.z * rs); a[15] = (__bf16)(a3.w * rs);
    // (neg_a, A, neg_b, B, c_mod, C, reuse_a, reuse_b)
    acc = __builtin_amdgcn_wmma_f32_16x16x32_bf16(false, a, false, b,
                                                  (short)0, acc, false, false);
  }

  int bcol = n0 + l15;
  if (bcol < N) {
    float bi = bias[bcol];
#pragma unroll
    for (int r = 0; r < 8; ++r) {
      int row = m0 + r + hi * 8;
      Y[(size_t)row * N + bcol] = acc[r] + bi;
    }
  }
}

// ------------------------------ launcher -----------------------------------

extern "C" void kernel_launch(void* const* d_in, const int* in_sizes, int n_in,
                              void* d_out, int out_size, void* d_ws, size_t ws_size,
                              hipStream_t stream) {
  (void)in_sizes; (void)n_in; (void)out_size; (void)ws_size;

  const float* fsnet    = (const float*)d_in[0];
  const int*   src      = (const int*)d_in[1];
  const int*   dst      = (const int*)d_in[2];
  const int*   graph_id = (const int*)d_in[3];
  const float* W_ext    = (const float*)d_in[4];
  const float* b_ext    = (const float*)d_in[5];
  const float* W1       = (const float*)d_in[6];
  const float* b1       = (const float*)d_in[7];
  const float* W2       = (const float*)d_in[8];
  const float* b2       = (const float*)d_in[9];
  const float* Wc       = (const float*)d_in[10];
  const float* bc       = (const float*)d_in[11];

  const int NN = 50000, NE = 800000, NG = 256, L = 100;
  const int AGG_LDA = 128;    // 100 padded to mult. of 32 (zero-filled)
  const int POOL_LDA = 224;   // 200 padded to mult. of 32 (zero-filled)

  // GEMM geometry: {Kchunks, N, Ntiles}
  const int KC1 = 8,  N1 = 100, NT1 = 7;   // fsnet[.,256] @ W_ext -> [.,100]
  const int KC2 = 4,  N2 = 100, NT2 = 7;   // agg(128)    @ W1    -> [.,100]
  const int KC3 = 4,  N3 = 200, NT3 = 13;  // agg(128)    @ W2    -> [.,200]
  const int KC4 = 7,  N4 = 55,  NT4 = 4;   // pooled(224) @ Wc    -> [.,55]

  // workspace layout (floats), ~87 MB total
  float* ws     = (float*)d_ws;
  float* hbuf   = ws;                               // [NN,100]  h0 then h1
  float* aggp   = hbuf   + (size_t)NN * L;          // [NN,128]  padded scatter target
  float* h2     = aggp   + (size_t)NN * AGG_LDA;    // [NN,200]
  float* degO   = h2     + (size_t)NN * 2 * L;      // [NN] -> inv_sqrt_out
  float* degI   = degO   + NN;                      // [NN] -> inv_sqrt_in
  float* pooled = degI   + NN;                      // [NG,224] padded
  float* cnt    = pooled + (size_t)NG * POOL_LDA;   // [NG] -> 1/count
  __bf16* bp1   = (__bf16*)(cnt + NG);              // packed B fragments
  __bf16* bp2   = bp1 + (size_t)NT1 * KC1 * 32 * 16;
  __bf16* bp3   = bp2 + (size_t)NT2 * KC2 * 32 * 16;
  __bf16* bp4   = bp3 + (size_t)NT3 * KC3 * 32 * 16;

  // --- pack all weight matrices into WMMA B-fragment layout (tiny) ---
  {
    int t1 = NT1 * KC1 * 32, t2 = NT2 * KC2 * 32, t3 = NT3 * KC3 * 32, t4 = NT4 * KC4 * 32;
    pack_B_kernel<<<(t1 + 127) / 128, 128, 0, stream>>>(W_ext, bp1, 256,     N1, KC1, t1);
    pack_B_kernel<<<(t2 + 127) / 128, 128, 0, stream>>>(W1,    bp2, L,       N2, KC2, t2);
    pack_B_kernel<<<(t3 + 127) / 128, 128, 0, stream>>>(W2,    bp3, L,       N3, KC3, t3);
    pack_B_kernel<<<(t4 + 127) / 128, 128, 0, stream>>>(Wc,    bp4, 2 * L,   N4, KC4, t4);
  }

  // --- degrees -> inv sqrt norms ---
  zero_f32<<<256, 256, 0, stream>>>(degO, 2L * NN);                         // degO+degI
  zero_f32<<<256, 256, 0, stream>>>(pooled, (long)NG * POOL_LDA + NG);      // pooled+cnt
  degree_kernel<<<(NE + 255) / 256, 256, 0, stream>>>(src, dst, degO, degI, NE);
  rsqrt_clip_kernel<<<(2 * NN + 255) / 256, 256, 0, stream>>>(degO, 2 * NN);

  // --- extractor: h0 = fsnet @ W_ext + b_ext ---
  {
    int tiles = (NN / 16) * NT1;
    gemm_bias_wmma_kernel<<<(tiles + 3) / 4, 128, 0, stream>>>(
        fsnet, nullptr, bp1, b_ext, hbuf, 256, KC1, N1, NT1, tiles);
  }

  // --- GCN layer 1 ---
  zero_f32<<<2048, 256, 0, stream>>>(aggp, (long)NN * AGG_LDA);
  edge_scatter_kernel<<<(NE * 25 + 255) / 256, 256, 0, stream>>>(
      hbuf, degO, src, dst, aggp, NE, AGG_LDA);
  {
    int tiles = (NN / 16) * NT2;
    gemm_bias_wmma_kernel<<<(tiles + 3) / 4, 128, 0, stream>>>(
        aggp, degI, bp2, b1, hbuf, AGG_LDA, KC2, N2, NT2, tiles);  // inv_sqrt_in fused
  }

  // --- GCN layer 2 ---
  zero_f32<<<2048, 256, 0, stream>>>(aggp, (long)NN * AGG_LDA);
  edge_scatter_kernel<<<(NE * 25 + 255) / 256, 256, 0, stream>>>(
      hbuf, degO, src, dst, aggp, NE, AGG_LDA);
  {
    int tiles = (NN / 16) * NT3;
    gemm_bias_wmma_kernel<<<(tiles + 3) / 4, 128, 0, stream>>>(
        aggp, degI, bp3, b2, h2, AGG_LDA, KC3, N3, NT3, tiles);    // inv_sqrt_in fused
  }

  // --- per-graph mean pool (sum + counts; 1/cnt fused into final GEMM) ---
  count_kernel<<<(NN + 255) / 256, 256, 0, stream>>>(graph_id, cnt, NN);
  pool_scatter_kernel<<<(NN * 50 + 255) / 256, 256, 0, stream>>>(
      h2, graph_id, pooled, NN, POOL_LDA);
  inv_count_kernel<<<1, 256, 0, stream>>>(cnt, NG);

  // --- classifier: out = (pooled / cnt) @ Wc + bc ---
  {
    int tiles = (NG / 16) * NT4;
    gemm_bias_wmma_kernel<<<(tiles + 3) / 4, 128, 0, stream>>>(
        pooled, cnt, bp4, bc, (float*)d_out, POOL_LDA, KC4, N4, NT4, tiles);
  }
}
